// StateUpdateOperator_52243982188947
// MI455X (gfx1250) — matrix operand order repaired and checked by reading, not compiled, not run on hardware
//
#include <hip/hip_runtime.h>
#include <hip/hip_bf16.h>

// ---------------- problem constants ----------------
static constexpr int B_    = 2;
static constexpr int T_    = 2048;
static constexpr int D_    = 1024;
static constexpr int H_    = 16;
static constexpr int HD_   = 64;
static constexpr int FF_   = 4096;
static constexpr int M_    = B_ * T_;   // 4096 rows of the activation matrix

// ---------------- WMMA types ----------------
typedef __attribute__((ext_vector_type(16))) __bf16 v16bf;
typedef __attribute__((ext_vector_type(8)))  __bf16 v8bf;
typedef __attribute__((ext_vector_type(8)))  float  v8f;

union Frag16 { v16bf v; v8bf h[2]; };

static __device__ __forceinline__ v8f wmma_bf16(v16bf a, v16bf b, v8f c) {
  // D = A(16x32 bf16) * B(32x16 bf16) + C(16x16 f32)
  return __builtin_amdgcn_wmma_f32_16x16x32_bf16(false, a, false, b, (short)0, c, false, false);
}

// ---------------- f32 [K,N] -> bf16 [N,K] (transpose + convert) ----------------
__global__ __launch_bounds__(256) void convt_kernel(const float* __restrict__ W,
                                                    __bf16* __restrict__ Wt,
                                                    int K, int N) {
  int idx = blockIdx.x * 256 + threadIdx.x;   // over K*N, coalesced read
  int k = idx / N;
  int n = idx - k * N;
  Wt[(size_t)n * K + k] = (__bf16)W[idx];
}

// ---------------- LayerNorm: f32 [M,1024] -> bf16 [M,1024] ----------------
__global__ __launch_bounds__(256) void layernorm_kernel(const float* __restrict__ x,
                                                        const float* __restrict__ g,
                                                        const float* __restrict__ bb,
                                                        __bf16* __restrict__ out) {
  int row = blockIdx.x;
  int t   = threadIdx.x;
  const float* xr = x + (size_t)row * D_;
  float v0 = xr[t], v1 = xr[t + 256], v2 = xr[t + 512], v3 = xr[t + 768];
  float s  = v0 + v1 + v2 + v3;
  float s2 = v0*v0 + v1*v1 + v2*v2 + v3*v3;
  #pragma unroll
  for (int off = 16; off; off >>= 1) {
    s  += __shfl_xor(s,  off, 32);
    s2 += __shfl_xor(s2, off, 32);
  }
  __shared__ float rs[8], rs2[8];
  __shared__ float smu, srinv;
  int wave = t >> 5, lane = t & 31;
  if (lane == 0) { rs[wave] = s; rs2[wave] = s2; }
  __syncthreads();
  if (t == 0) {
    float S = 0.f, S2 = 0.f;
    #pragma unroll
    for (int w = 0; w < 8; w++) { S += rs[w]; S2 += rs2[w]; }
    float mu  = S * (1.0f / D_);
    float var = S2 * (1.0f / D_) - mu * mu;
    smu = mu; srinv = rsqrtf(var + 1e-5f);
  }
  __syncthreads();
  float mu = smu, rinv = srinv;
  __bf16* orow = out + (size_t)row * D_;
  orow[t]       = (__bf16)((v0 - mu) * rinv * g[t]       + bb[t]);
  orow[t + 256] = (__bf16)((v1 - mu) * rinv * g[t + 256] + bb[t + 256]);
  orow[t + 512] = (__bf16)((v2 - mu) * rinv * g[t + 512] + bb[t + 512]);
  orow[t + 768] = (__bf16)((v3 - mu) * rinv * g[t + 768] + bb[t + 768]);
}

// ---------------- WMMA GEMM: C[M,N] = epi(A_bf16[M,K] @ Bt_bf16[N,K]^T + bias) ----------------
// 256 threads = 8 waves; block tile 128x128; wave tile 32(M) x 64(N); K step 32.
__global__ __launch_bounds__(256) void gemm_bf16_kernel(
    const __bf16* __restrict__ A, const __bf16* __restrict__ Bt,
    const float* __restrict__ bias, const float* __restrict__ resid,
    float* __restrict__ outF, __bf16* __restrict__ outB,
    int M, int N, int K, int doGelu) {
  int wave = threadIdx.x >> 5;
  int lane = threadIdx.x & 31;
  int lh = lane >> 4, ln = lane & 15;
  int m0 = blockIdx.y * 128 + (wave >> 1) * 32;
  int n0 = blockIdx.x * 128 + (wave &  1) * 64;

  v8f C[2][4];
  #pragma unroll
  for (int i = 0; i < 2; i++)
    #pragma unroll
    for (int j = 0; j < 4; j++)
      #pragma unroll
      for (int e = 0; e < 8; e++) C[i][j][e] = 0.0f;

  for (int kk = 0; kk < K; kk += 32) {
    Frag16 a[2];
    #pragma unroll
    for (int i = 0; i < 2; i++) {
      const __bf16* ap = A + (size_t)(m0 + i * 16 + ln) * K + kk;
      __builtin_prefetch(ap + 32, 0, 1);               // next K tile -> global_prefetch
      a[i].h[0] = *(const v8bf*)(ap + lh * 16);        // VGPR0-3 half
      a[i].h[1] = *(const v8bf*)(ap + lh * 16 + 8);    // VGPR4-7 half
    }
    #pragma unroll
    for (int j = 0; j < 4; j++) {
      const __bf16* bp = Bt + (size_t)(n0 + j * 16 + ln) * K + kk + lh * 16;
      v16bf bfrag = *(const v16bf*)bp;                 // 32B contiguous per lane
      #pragma unroll
      for (int i = 0; i < 2; i++)
        C[i][j] = wmma_bf16(a[i].v, bfrag, C[i][j]);
    }
  }

  #pragma unroll
  for (int i = 0; i < 2; i++)
    #pragma unroll
    for (int j = 0; j < 4; j++) {
      int col = n0 + j * 16 + ln;
      float bcol = bias[col];
      #pragma unroll
      for (int v = 0; v < 8; v++) {
        int row = m0 + i * 16 + v + 8 * lh;            // C-frag row mapping
        float val = C[i][j][v] + bcol;
        if (resid) val += resid[(size_t)row * N + col];
        if (doGelu) val = 0.5f * val * (1.0f + erff(val * 0.70710678118654752f));
        if (outB) outB[(size_t)row * N + col] = (__bf16)val;
        else      outF[(size_t)row * N + col] = val;
      }
    }
}

// ---------------- RoPE + layout: Qf/Kf/Vf bf16 [M,D] -> Q,K [B,H,T,64], Vt [B,H,64,T] ----------------
__global__ __launch_bounds__(256) void rope_kernel(const __bf16* __restrict__ Qf,
                                                   const __bf16* __restrict__ Kf,
                                                   const __bf16* __restrict__ Vf,
                                                   __bf16* __restrict__ Qo,
                                                   __bf16* __restrict__ Ko,
                                                   __bf16* __restrict__ Vto) {
  int gid = blockIdx.x * 256 + threadIdx.x;            // b*T*D + t*D + h*64 + hd
  int hd = gid & 63;
  int h  = (gid >> 6) & 15;
  int t  = (gid >> 10) & (T_ - 1);
  int b  = gid >> 21;
  float q = (float)Qf[gid];
  float k = (float)Kf[gid];
  int j = hd & 31;
  float ang = (float)t * __expf(-(float)j * (9.210340371976184f / 32.0f)); // 10000^(-2j/64)
  float sn, cs;
  __sincosf(ang, &sn, &cs);
  float qr, kr;
  if (hd < 32) { qr = -(float)Qf[gid + 32]; kr = -(float)Kf[gid + 32]; }
  else         { qr =  (float)Qf[gid - 32]; kr =  (float)Kf[gid - 32]; }
  size_t oIdx = (((size_t)(b * H_ + h) * T_ + t) * HD_) + hd;
  Qo[oIdx] = (__bf16)(q * cs + qr * sn);
  Ko[oIdx] = (__bf16)(k * cs + kr * sn);
  Vto[(((size_t)(b * H_ + h) * HD_ + hd) * T_) + t] = Vf[gid];
}

// ---------------- Flash attention (WMMA), one wave = 16 query rows ----------------
__global__ __launch_bounds__(256) void flash_attn_kernel(
    const __bf16* __restrict__ Q, const __bf16* __restrict__ Kx,
    const __bf16* __restrict__ Vt, const float* __restrict__ mask,
    __bf16* __restrict__ Oout) {
  __shared__ __bf16 Plds[8][16 * 32];                  // per-wave P staging (C->A layout)
  int wave = threadIdx.x >> 5;
  int lane = threadIdx.x & 31;
  int lh = lane >> 4, ln = lane & 15;
  int bh = blockIdx.x;                                 // b*H + h
  int b  = bh >> 4;
  int h  = bh & 15;
  int q0 = blockIdx.y * 128 + wave * 16;

  const __bf16* Qp = Q  + (size_t)bh * T_ * HD_;
  const __bf16* Kp = Kx + (size_t)bh * T_ * HD_;
  const __bf16* Vp = Vt + (size_t)bh * HD_ * T_;

  Frag16 qa0, qa1;                                     // Q 16x64 as two 16x32 A-frags
  {
    const __bf16* qr = Qp + (size_t)(q0 + ln) * HD_;
    qa0.h[0] = *(const v8bf*)(qr + lh * 16);
    qa0.h[1] = *(const v8bf*)(qr + 8 + lh * 16);
    qa1.h[0] = *(const v8bf*)(qr + 32 + lh * 16);
    qa1.h[1] = *(const v8bf*)(qr + 40 + lh * 16);
  }

  v8f O[4];
  #pragma unroll
  for (int c = 0; c < 4; c++)
    #pragma unroll
    for (int e = 0; e < 8; e++) O[c][e] = 0.0f;
  float rowMax[8], rowSum[8];
  #pragma unroll
  for (int v = 0; v < 8; v++) { rowMax[v] = -3.0e38f; rowSum[v] = 0.0f; }

  __bf16* P = &Plds[wave][0];

  for (int kt = 0; kt < T_; kt += 32) {
    // K^T B-frags: contraction = head dim, columns = keys
    const __bf16* kr0 = Kp + (size_t)(kt + ln) * HD_;
    const __bf16* kr1 = Kp + (size_t)(kt + 16 + ln) * HD_;
    v16bf kb00 = *(const v16bf*)(kr0 + lh * 16);
    v16bf kb01 = *(const v16bf*)(kr0 + 32 + lh * 16);
    v16bf kb10 = *(const v16bf*)(kr1 + lh * 16);
    v16bf kb11 = *(const v16bf*)(kr1 + 32 + lh * 16);

    v8f S0, S1;
    #pragma unroll
    for (int e = 0; e < 8; e++) { S0[e] = 0.0f; S1[e] = 0.0f; }
    S0 = wmma_bf16(qa0.v, kb00, S0);
    S0 = wmma_bf16(qa1.v, kb01, S0);
    S1 = wmma_bf16(qa0.v, kb10, S1);
    S1 = wmma_bf16(qa1.v, kb11, S1);

    #pragma unroll
    for (int v = 0; v < 8; v++) {
      int qq = q0 + v + 8 * lh;
      const float* mrow = mask + (size_t)qq * T_ + kt;
      float s0 = S0[v] * 0.125f + mrow[ln];            // 1/sqrt(64)
      float s1 = S1[v] * 0.125f + mrow[16 + ln];
      float mx = fmaxf(s0, s1);
      #pragma unroll
      for (int off = 8; off; off >>= 1) mx = fmaxf(mx, __shfl_xor(mx, off, 16));
      mx = fmaxf(mx, rowMax[v]);
      float corr = __expf(rowMax[v] - mx);
      rowMax[v] = mx;
      float p0 = __expf(s0 - mx);
      float p1 = __expf(s1 - mx);
      float ss = p0 + p1;
      #pragma unroll
      for (int off = 8; off; off >>= 1) ss += __shfl_xor(ss, off, 16);
      rowSum[v] = rowSum[v] * corr + ss;
      O[0][v] *= corr; O[1][v] *= corr; O[2][v] *= corr; O[3][v] *= corr;
      int m = v + 8 * lh;
      P[m * 32 + ln]      = (__bf16)p0;                // C layout -> row-major LDS
      P[m * 32 + 16 + ln] = (__bf16)p1;
    }

    // reload P in A-frag layout (same-wave LDS ops stay in order; waits inserted by compiler)
    Frag16 pa;
    pa.h[0] = *(const v8bf*)(P + ln * 32 + lh * 16);
    pa.h[1] = *(const v8bf*)(P + ln * 32 + lh * 16 + 8);

    #pragma unroll
    for (int c = 0; c < 4; c++) {
      v16bf vb = *(const v16bf*)(Vp + (size_t)(c * 16 + ln) * T_ + kt + lh * 16);
      O[c] = wmma_bf16(pa.v, vb, O[c]);
    }
  }

  #pragma unroll
  for (int v = 0; v < 8; v++) {
    float inv = 1.0f / rowSum[v];
    int row = q0 + v + 8 * lh;
    __bf16* orow = Oout + ((size_t)(b * T_ + row)) * D_ + h * HD_;
    #pragma unroll
    for (int c = 0; c < 4; c++)
      orow[c * 16 + ln] = (__bf16)(O[c][v] * inv);
  }
}

// ---------------- host orchestration ----------------
extern "C" void kernel_launch(void* const* d_in, const int* in_sizes, int n_in,
                              void* d_out, int out_size, void* d_ws, size_t ws_size,
                              hipStream_t stream) {
  (void)in_sizes; (void)n_in; (void)out_size; (void)ws_size;
  const float* x    = (const float*)d_in[0];
  const float* mask = (const float*)d_in[1];
  const float* ln1g = (const float*)d_in[2];
  const float* ln1b = (const float*)d_in[3];
  const float* Wq   = (const float*)d_in[4];
  const float* bq   = (const float*)d_in[5];
  const float* Wk   = (const float*)d_in[6];
  const float* bk   = (const float*)d_in[7];
  const float* Wv   = (const float*)d_in[8];
  const float* bv   = (const float*)d_in[9];
  const float* Wo   = (const float*)d_in[10];
  const float* bo   = (const float*)d_in[11];
  const float* ln2g = (const float*)d_in[12];
  const float* ln2b = (const float*)d_in[13];
  const float* W1   = (const float*)d_in[14];
  const float* b1   = (const float*)d_in[15];
  const float* W2   = (const float*)d_in[16];
  const float* b2   = (const float*)d_in[17];
  float* out = (float*)d_out;

  char* ws = (char*)d_ws;
  size_t off = 0;
  auto take = [&](size_t bytes) -> char* {
    char* p = ws + off;
    off += (bytes + 255) & ~(size_t)255;
    return p;
  };
  __bf16* normed1 = (__bf16*)take((size_t)M_ * D_ * 2);
  __bf16* WqT = (__bf16*)take((size_t)D_ * D_ * 2);
  __bf16* WkT = (__bf16*)take((size_t)D_ * D_ * 2);
  __bf16* WvT = (__bf16*)take((size_t)D_ * D_ * 2);
  __bf16* WoT = (__bf16*)take((size_t)D_ * D_ * 2);
  __bf16* W1T = (__bf16*)take((size_t)FF_ * D_ * 2);
  __bf16* W2T = (__bf16*)take((size_t)D_ * FF_ * 2);
  char*   R   = take((size_t)M_ * FF_ * 2);            // 32MB region, reused
  __bf16* Qf  = (__bf16*)R;                            // [M,D] pre-RoPE
  __bf16* Kf  = (__bf16*)(R + (size_t)M_ * D_ * 2);
  __bf16* Vf  = (__bf16*)(R + (size_t)2 * M_ * D_ * 2);
  __bf16* h1  = (__bf16*)R;                            // FFN hidden, aliases dead Qf/Kf/Vf
  __bf16* Qr  = (__bf16*)take((size_t)M_ * HD_ * H_ * 2);  // [B,H,T,64]
  __bf16* Kr  = (__bf16*)take((size_t)M_ * HD_ * H_ * 2);
  __bf16* VtT = (__bf16*)take((size_t)M_ * HD_ * H_ * 2);  // [B,H,64,T]
  __bf16* attnO = (__bf16*)take((size_t)M_ * D_ * 2);
  float*  x2  = (float*)take((size_t)M_ * D_ * 4);
  __bf16* normed2 = (__bf16*)take((size_t)M_ * D_ * 2);

  dim3 blk(256);
  // weight convert + transpose (f32 [K,N] -> bf16 [N,K])
  convt_kernel<<<(D_ * D_) / 256, blk, 0, stream>>>(Wq, WqT, D_, D_);
  convt_kernel<<<(D_ * D_) / 256, blk, 0, stream>>>(Wk, WkT, D_, D_);
  convt_kernel<<<(D_ * D_) / 256, blk, 0, stream>>>(Wv, WvT, D_, D_);
  convt_kernel<<<(D_ * D_) / 256, blk, 0, stream>>>(Wo, WoT, D_, D_);
  convt_kernel<<<(D_ * FF_) / 256, blk, 0, stream>>>(W1, W1T, D_, FF_);
  convt_kernel<<<(FF_ * D_) / 256, blk, 0, stream>>>(W2, W2T, FF_, D_);

  // LN1
  layernorm_kernel<<<M_, blk, 0, stream>>>(x, ln1g, ln1b, normed1);

  // QKV projections (bf16 output)
  dim3 g1(D_ / 128, M_ / 128);
  gemm_bf16_kernel<<<g1, blk, 0, stream>>>(normed1, WqT, bq, nullptr, nullptr, Qf, M_, D_, D_, 0);
  gemm_bf16_kernel<<<g1, blk, 0, stream>>>(normed1, WkT, bk, nullptr, nullptr, Kf, M_, D_, D_, 0);
  gemm_bf16_kernel<<<g1, blk, 0, stream>>>(normed1, WvT, bv, nullptr, nullptr, Vf, M_, D_, D_, 0);

  // RoPE + head-major layout (+ V transpose)
  rope_kernel<<<(M_ * D_) / 256, blk, 0, stream>>>(Qf, Kf, Vf, Qr, Kr, VtT);

  // flash attention
  dim3 ga(B_ * H_, T_ / 128);
  flash_attn_kernel<<<ga, blk, 0, stream>>>(Qr, Kr, VtT, mask, attnO);

  // output projection + residual (f32)
  gemm_bf16_kernel<<<g1, blk, 0, stream>>>(attnO, WoT, bo, x, x2, nullptr, M_, D_, D_, 0);

  // LN2
  layernorm_kernel<<<M_, blk, 0, stream>>>(x2, ln2g, ln2b, normed2);

  // FFN1 + exact GELU (bf16 output)
  dim3 g2(FF_ / 128, M_ / 128);
  gemm_bf16_kernel<<<g2, blk, 0, stream>>>(normed2, W1T, b1, nullptr, nullptr, h1, M_, FF_, D_, 1);

  // FFN2 + residual -> final output
  gemm_bf16_kernel<<<g1, blk, 0, stream>>>(h1, W2T, b2, x2, out, nullptr, M_, D_, FF_, 0);
}
